// MultiHeadAttention_91096256348379
// MI455X (gfx1250) — matrix operand order repaired and checked by reading, not compiled, not run on hardware
//
#include <hip/hip_runtime.h>
#include <hip/hip_bf16.h>

#define HIDDEN   1024
#define HEADS    16
#define HEAD_DIM 64
#define SEQ      2048
#define BATCH    4
#define NTOK     (BATCH * SEQ)   // 8192

typedef __bf16 bf16_t;
typedef __bf16 v16bf __attribute__((ext_vector_type(16)));
typedef __bf16 v8bf  __attribute__((ext_vector_type(8)));
typedef __bf16 v4bf  __attribute__((ext_vector_type(4)));
typedef float  v8f   __attribute__((ext_vector_type(8)));
typedef float  v4f   __attribute__((ext_vector_type(4)));

static __device__ __forceinline__ v16bf combine8(v8bf lo, v8bf hi) {
    v16bf r;
#pragma unroll
    for (int i = 0; i < 8; ++i) { r[i] = lo[i]; r[i + 8] = hi[i]; }
    return r;
}

static __device__ __forceinline__ v8f wmma_bf16(v16bf a, v16bf b, v8f c) {
    return __builtin_amdgcn_wmma_f32_16x16x32_bf16(
        /*neg_a=*/false, a, /*neg_b=*/false, b,
        /*c_mod=*/(short)0, c, /*reuse_a=*/false, /*reuse_b=*/false);
}

// raw 32-bit LDS offset (the ISA "dsaddr") from a __shared__ pointer
static __device__ __forceinline__ unsigned lds_off32(const void* p) {
    return (unsigned)(size_t)(__attribute__((address_space(3))) const char*)p;
}

// async DMA: 16 bytes per lane, global -> LDS, tracked by ASYNCcnt
static __device__ __forceinline__ void async_ld_b128(unsigned lds, const void* gp) {
    asm volatile("global_load_async_to_lds_b128 %0, %1, off"
                 :: "v"(lds), "v"(gp) : "memory");
}
static __device__ __forceinline__ void wait_async0() {
    asm volatile("s_wait_asynccnt 0x0" ::: "memory");
}

// ---------------------------------------------------------------- fp32 -> bf16
__global__ __launch_bounds__(256)
void cvt_f32_bf16(const float* __restrict__ src, bf16_t* __restrict__ dst, int n4) {
    int i = blockIdx.x * blockDim.x + threadIdx.x;
    if (i < n4) {
        v4f v = ((const v4f*)src)[i];
        v4bf o;
#pragma unroll
        for (int j = 0; j < 4; ++j) o[j] = (bf16_t)v[j];
        ((v4bf*)dst)[i] = o;
    }
}

// ---------------------------------------------------------------- GEMM: Y = X @ W^T + b
// A: [M, 1024] bf16 row-major.  W: [1024, 1024] bf16 row-major (N x K).
// mode 0: bf16 scatter head-major [b,h,s,d], val = (acc+bias)*scale   (Q scale=1/8, K scale=1)
// mode 2: bf16 scatter transposed [b,h,d,s]                           (V)
// mode 3: fp32 row-major [M,N] + bias                                 (final out-proj)
#define GBM 64
#define GBN 128
#define GBK 32
#define GLDA 40   // padded LDS k-stride (elements): 80B rows, 16B-aligned chunks

__global__ __launch_bounds__(256)
void gemm_xwT(const bf16_t* __restrict__ A, const bf16_t* __restrict__ W,
              const float* __restrict__ bias, void* __restrict__ dst,
              int mode, float scale) {
    __shared__ bf16_t sA[2][GBM * GLDA];
    __shared__ bf16_t sB[2][GBN * GLDA];
    const int K = HIDDEN;
    const int bm = blockIdx.x * GBM;
    const int bn = blockIdx.y * GBN;
    const int t = threadIdx.x;
    const int wave = t >> 5;
    const int lane = t & 31;
    const int lr = lane & 15;
    const bool hihalf = lane >= 16;
    const int wm = (wave & 1) * 32;   // wave M offset inside WG tile
    const int wn = (wave >> 1) * 32;  // wave N offset inside WG tile

    v8f acc[2][2];
#pragma unroll
    for (int i = 0; i < 2; ++i)
#pragma unroll
        for (int j = 0; j < 2; ++j) acc[i][j] = (v8f)0.0f;

    // per-thread DMA slots: A = one 16B chunk, B = two 16B chunks
    const int arow = t >> 2,        akc  = (t & 3) * 8;
    const int b0i = t * 2,          b1i  = t * 2 + 1;
    const int b0row = b0i >> 2,     b0kc = (b0i & 3) * 8;
    const int b1row = b1i >> 2,     b1kc = (b1i & 3) * 8;

    unsigned aoff[2], b0off[2], b1off[2];
#pragma unroll
    for (int bi = 0; bi < 2; ++bi) {
        aoff[bi]  = lds_off32(&sA[bi][arow * GLDA + akc]);
        b0off[bi] = lds_off32(&sB[bi][b0row * GLDA + b0kc]);
        b1off[bi] = lds_off32(&sB[bi][b1row * GLDA + b1kc]);
    }
    const bf16_t* aptr  = &A[(size_t)(bm + arow) * K + akc];
    const bf16_t* bptr0 = &W[(size_t)(bn + b0row) * K + b0kc];
    const bf16_t* bptr1 = &W[(size_t)(bn + b1row) * K + b1kc];

    // prologue: DMA first k-slab into buffer 0
    async_ld_b128(aoff[0],  aptr);
    async_ld_b128(b0off[0], bptr0);
    async_ld_b128(b1off[0], bptr1);

    int buf = 0;
    for (int k0 = 0; k0 < K; k0 += GBK) {
        wait_async0();       // my writes into sX[buf] landed
        __syncthreads();     // everyone's landed; nobody still reads sX[buf^1]
        if (k0 + GBK < K) {  // overlap next slab DMA with this slab's WMMAs
            async_ld_b128(aoff[buf ^ 1],  aptr  + k0 + GBK);
            async_ld_b128(b0off[buf ^ 1], bptr0 + k0 + GBK);
            async_ld_b128(b1off[buf ^ 1], bptr1 + k0 + GBK);
        }
        const bf16_t* cA = &sA[buf][0];
        const bf16_t* cB = &sB[buf][0];

        // A fragments: lanes 0-15 hold K 0..15, lanes 16-31 hold K 16..31
        v16bf af[2], bfr[2];
        const int ac0 = hihalf ? 16 : 0;
#pragma unroll
        for (int mt = 0; mt < 2; ++mt) {
            int row = wm + mt * 16 + lr;
            af[mt] = combine8(*(const v8bf*)&cA[row * GLDA + ac0],
                              *(const v8bf*)&cA[row * GLDA + ac0 + 8]);
        }
        // B fragments: lane n = column; lanes 0-15 K 0..15, lanes 16-31 K 16..31
        const int bc0 = hihalf ? 16 : 0;
#pragma unroll
        for (int nt = 0; nt < 2; ++nt) {
            int col = wn + nt * 16 + lr;
            bfr[nt] = combine8(*(const v8bf*)&cB[col * GLDA + bc0],
                               *(const v8bf*)&cB[col * GLDA + bc0 + 8]);
        }
#pragma unroll
        for (int mt = 0; mt < 2; ++mt)
#pragma unroll
            for (int nt = 0; nt < 2; ++nt)
                acc[mt][nt] = wmma_bf16(af[mt], bfr[nt], acc[mt][nt]);
        buf ^= 1;
    }

    // epilogue: C layout = VGPR r, lanes 0-15 -> M=r, lanes 16-31 -> M=8+r
#pragma unroll
    for (int mt = 0; mt < 2; ++mt) {
#pragma unroll
        for (int nt = 0; nt < 2; ++nt) {
#pragma unroll
            for (int r = 0; r < 8; ++r) {
                int mrow = bm + wm + mt * 16 + r + (hihalf ? 8 : 0);
                int ncol = bn + wn + nt * 16 + lr;
                float v = (acc[mt][nt][r] + bias[ncol]) * scale;
                if (mode == 3) {
                    ((float*)dst)[(size_t)mrow * HIDDEN + ncol] = v;
                } else {
                    int b = mrow >> 11;         // token / 2048
                    int s = mrow & (SEQ - 1);
                    int h = ncol >> 6;
                    int d = ncol & (HEAD_DIM - 1);
                    size_t bh = (size_t)(b * HEADS + h);
                    if (mode == 2)  // V transposed [b,h,d,s]
                        ((bf16_t*)dst)[(bh * HEAD_DIM + d) * SEQ + s] = (bf16_t)v;
                    else            // Q/K head-major [b,h,s,d]
                        ((bf16_t*)dst)[(bh * SEQ + s) * HEAD_DIM + d] = (bf16_t)v;
                }
            }
        }
    }
}

// ---------------------------------------------------------------- flash attention
// Q pre-scaled by 1/sqrt(d). 4 waves/WG, 16 q-rows per wave, 32-key blocks.
#define KB 32

__global__ __launch_bounds__(128)
void flash_attn(const bf16_t* __restrict__ Qh, const bf16_t* __restrict__ Kh,
                const bf16_t* __restrict__ Vt, const int* __restrict__ amask,
                bf16_t* __restrict__ outp) {
    __shared__ bf16_t Plds[4][16][KB];   // per-wave P staging (C-layout -> A-layout)
    const int t = threadIdx.x;
    const int wave = t >> 5, lane = t & 31, lr = lane & 15;
    const bool hihalf = lane >= 16;
    const int nqb = SEQ / 64;
    const int bh = blockIdx.x / nqb;
    const int qb = blockIdx.x % nqb;
    const int b = bh / HEADS, h = bh % HEADS;

    const bf16_t* Q  = Qh + (size_t)bh * SEQ * HEAD_DIM;
    const bf16_t* Kp = Kh + (size_t)bh * SEQ * HEAD_DIM;
    const bf16_t* V  = Vt + (size_t)bh * HEAD_DIM * SEQ;
    const int* mrow = amask + b * SEQ;
    const int q0 = qb * 64 + wave * 16;
    const float NEGINF = -__builtin_inff();

    // Q A-fragments for the two 32-wide d chunks
    v16bf qf[2];
    const int ac0 = hihalf ? 16 : 0;
#pragma unroll
    for (int kd = 0; kd < 2; ++kd) {
        int row = q0 + lr;
        int d0 = kd * 32 + ac0;
        qf[kd] = combine8(*(const v8bf*)&Q[(size_t)row * HEAD_DIM + d0],
                          *(const v8bf*)&Q[(size_t)row * HEAD_DIM + d0 + 8]);
    }

    v8f acc[4];
#pragma unroll
    for (int i = 0; i < 4; ++i) acc[i] = (v8f)0.0f;
    float rmax[8], rsum[8];
#pragma unroll
    for (int r = 0; r < 8; ++r) { rmax[r] = NEGINF; rsum[r] = 0.0f; }

    for (int j = 0; j < SEQ; j += KB) {
        // K B-fragments: lane = key column
        v16bf kf[2][2];
        const int bc0 = hihalf ? 16 : 0;
#pragma unroll
        for (int nt = 0; nt < 2; ++nt) {
            int key = j + nt * 16 + lr;
#pragma unroll
            for (int kd = 0; kd < 2; ++kd) {
                int d0 = kd * 32 + bc0;
                kf[nt][kd] = combine8(*(const v8bf*)&Kp[(size_t)key * HEAD_DIM + d0],
                                      *(const v8bf*)&Kp[(size_t)key * HEAD_DIM + d0 + 8]);
            }
        }
        v8f s0 = (v8f)0.0f, s1 = (v8f)0.0f;
#pragma unroll
        for (int kd = 0; kd < 2; ++kd) {
            s0 = wmma_bf16(qf[kd], kf[0][kd], s0);
            s1 = wmma_bf16(qf[kd], kf[1][kd], s1);
        }

        // column mask, branchless: masked column -> score + (-inf) = -inf
        float madd0 = (mrow[j + lr]      == 0) ? NEGINF : 0.0f;
        float madd1 = (mrow[j + 16 + lr] == 0) ? NEGINF : 0.0f;
#pragma unroll
        for (int r = 0; r < 8; ++r) { s0[r] += madd0; s1[r] += madd1; }

        // online softmax per row (row lives in 16-lane half, per VGPR r)
        float p0[8], p1[8];
#pragma unroll
        for (int r = 0; r < 8; ++r) {
            float bm = fmaxf(s0[r], s1[r]);
            bm = fmaxf(bm, __shfl_xor(bm, 1));
            bm = fmaxf(bm, __shfl_xor(bm, 2));
            bm = fmaxf(bm, __shfl_xor(bm, 4));
            bm = fmaxf(bm, __shfl_xor(bm, 8));
            float nm = fmaxf(rmax[r], bm);
            float corr = __expf(fmaxf(rmax[r] - nm, -80.0f));
            rmax[r] = nm;
            p0[r] = __expf(fmaxf(s0[r] - nm, -80.0f));
            p1[r] = __expf(fmaxf(s1[r] - nm, -80.0f));
            float bs = p0[r] + p1[r];
            bs += __shfl_xor(bs, 1);
            bs += __shfl_xor(bs, 2);
            bs += __shfl_xor(bs, 4);
            bs += __shfl_xor(bs, 8);
            rsum[r] = rsum[r] * corr + bs;
#pragma unroll
            for (int dt = 0; dt < 4; ++dt) acc[dt][r] *= corr;
        }

        // transpose P through LDS: C-layout -> A-layout
#pragma unroll
        for (int r = 0; r < 8; ++r) {
            int rl = r + (hihalf ? 8 : 0);
            Plds[wave][rl][lr]      = (bf16_t)p0[r];
            Plds[wave][rl][16 + lr] = (bf16_t)p1[r];
        }
        __syncthreads();
        v16bf pf = combine8(*(const v8bf*)&Plds[wave][lr][ac0],
                            *(const v8bf*)&Plds[wave][lr][ac0 + 8]);

        // PV: B = V^T fragments, lane = output column d, contraction = 32 keys
#pragma unroll
        for (int dt = 0; dt < 4; ++dt) {
            int d = dt * 16 + lr;
            int kk = j + (hihalf ? 16 : 0);
            v16bf vf = combine8(*(const v8bf*)&V[(size_t)d * SEQ + kk],
                                *(const v8bf*)&V[(size_t)d * SEQ + kk + 8]);
            acc[dt] = wmma_bf16(pf, vf, acc[dt]);
        }
        __syncthreads();
    }

    // normalize and write token-major bf16 [b,s,h*64+d]
#pragma unroll
    for (int r = 0; r < 8; ++r) {
        float inv = rsum[r] > 0.0f ? 1.0f / rsum[r] : 0.0f;
        int rl = r + (hihalf ? 8 : 0);
        int s = q0 + rl;
#pragma unroll
        for (int dt = 0; dt < 4; ++dt) {
            int d = dt * 16 + lr;
            outp[((size_t)(b * SEQ + s)) * HIDDEN + h * HEAD_DIM + d] =
                (bf16_t)(acc[dt][r] * inv);
        }
    }
}

// ---------------------------------------------------------------- launch
extern "C" void kernel_launch(void* const* d_in, const int* in_sizes, int n_in,
                              void* d_out, int out_size, void* d_ws, size_t ws_size,
                              hipStream_t stream) {
    const float* x     = (const float*)d_in[0];
    const int*   amask = (const int*)d_in[1];
    const float* wq = (const float*)d_in[2]; const float* bq = (const float*)d_in[3];
    const float* wk = (const float*)d_in[4]; const float* bk = (const float*)d_in[5];
    const float* wv = (const float*)d_in[6]; const float* bv = (const float*)d_in[7];
    const float* wo = (const float*)d_in[8]; const float* bo = (const float*)d_in[9];
    float* out = (float*)d_out;

    char* ws = (char*)d_ws;
    const size_t MB = 1u << 20;
    bf16_t* xb   = (bf16_t*)(ws + 0 * MB);    // 16 MB: x bf16
    bf16_t* wqb  = (bf16_t*)(ws + 16 * MB);   // 2 MB each
    bf16_t* wkb  = (bf16_t*)(ws + 18 * MB);
    bf16_t* wvb  = (bf16_t*)(ws + 20 * MB);
    bf16_t* wob  = (bf16_t*)(ws + 22 * MB);
    bf16_t* Qh   = (bf16_t*)(ws + 24 * MB);   // 16 MB [b,h,s,d] (pre-scaled)
    bf16_t* Kh   = (bf16_t*)(ws + 40 * MB);   // 16 MB [b,h,s,d]
    bf16_t* Vt   = (bf16_t*)(ws + 56 * MB);   // 16 MB [b,h,d,s]
    bf16_t* attn = (bf16_t*)(ws + 72 * MB);   // 16 MB [b,s,H]

    // fp32 -> bf16 conversions
    int nx4 = NTOK * HIDDEN / 4;
    cvt_f32_bf16<<<(nx4 + 255) / 256, 256, 0, stream>>>(x, xb, nx4);
    int nw4 = HIDDEN * HIDDEN / 4;
    cvt_f32_bf16<<<(nw4 + 255) / 256, 256, 0, stream>>>(wq, wqb, nw4);
    cvt_f32_bf16<<<(nw4 + 255) / 256, 256, 0, stream>>>(wk, wkb, nw4);
    cvt_f32_bf16<<<(nw4 + 255) / 256, 256, 0, stream>>>(wv, wvb, nw4);
    cvt_f32_bf16<<<(nw4 + 255) / 256, 256, 0, stream>>>(wo, wob, nw4);

    // QKV projections (scale 1/sqrt(64) folded into Q)
    dim3 gg(NTOK / GBM, HIDDEN / GBN);
    gemm_xwT<<<gg, 256, 0, stream>>>(xb, wqb, bq, Qh, /*mode=*/0, 0.125f);
    gemm_xwT<<<gg, 256, 0, stream>>>(xb, wkb, bk, Kh, /*mode=*/0, 1.0f);
    gemm_xwT<<<gg, 256, 0, stream>>>(xb, wvb, bv, Vt, /*mode=*/2, 1.0f);

    // flash attention
    flash_attn<<<BATCH * HEADS * (SEQ / 64), 128, 0, stream>>>(Qh, Kh, Vt, amask, attn);

    // output projection -> fp32
    gemm_xwT<<<gg, 256, 0, stream>>>(attn, wob, bo, out, /*mode=*/3, 1.0f);
}
